// MultiQueryAttention_16844861735371
// MI455X (gfx1250) — matrix operand order repaired and checked by reading, not compiled, not run on hardware
//
#include <hip/hip_runtime.h>

// ---------------------------------------------------------------------------
// MultiQueryAttention for MI455X (gfx1250): bf16 WMMA flash-attention with
// TDM (tensor_load_to_lds) KV-tile staging.
//   x[2,4096,512] -> K/V proj, 8 Q heads, flash attn, out proj -> [2,4096,64]
// ---------------------------------------------------------------------------

typedef __attribute__((ext_vector_type(16))) __bf16 v16bf;
typedef __attribute__((ext_vector_type(8)))  float  v8f;
typedef __attribute__((ext_vector_type(4)))  unsigned int v4u;
typedef __attribute__((ext_vector_type(8)))  int    v8i;
typedef __attribute__((ext_vector_type(4)))  int    v4i;

typedef unsigned short u16;
typedef unsigned int   u32;
typedef unsigned long long u64;

#define BB 2
#define SS 4096
#define WW 512
#define DD 64
#define HH 8
#define BS (BB * SS)            // 8192 rows
#define NQKV (DD + DD + HH*DD)  // 640 fused projection columns

union BF16x16 { v16bf v; uint4 q[2]; };

static __device__ __forceinline__ u16 f2bf(float f) {
  u32 u = __float_as_uint(f);
  u32 r = u + 0x7FFFu + ((u >> 16) & 1u);   // round-to-nearest-even
  return (u16)(r >> 16);
}

// A fragment (16x32 bf16) from a row-major [16][stride] source.
// lane<16: row=lane, K = kk+0..7, kk+16..23 ; lane>=16: row=lane-16, K = kk+8..15, kk+24..31
static __device__ __forceinline__ v16bf load_a16x32(const u16* base, int stride, int lane, int kk) {
  BF16x16 t;
  const int hf = lane >> 4;
  const int m  = lane & 15;
  const u16* p = base + m * stride + kk + hf * 8;
  t.q[0] = *(const uint4*)(p);
  t.q[1] = *(const uint4*)(p + 16);
  return t.v;
}

// B fragment (32x16 bf16) from a column-major [n][stride] source (K contiguous per column).
// lane<16: N=lane, K = kk+0..7, kk+16..23 ; lane>=16: N=lane-16, K = kk+8..15, kk+24..31
static __device__ __forceinline__ v16bf load_b32x16(const u16* base, int stride, int lane, int kk) {
  BF16x16 t;
  const int hf = lane >> 4;
  const int n  = lane & 15;
  const u16* p = base + n * stride + kk + hf * 8;
  t.q[0] = *(const uint4*)(p);
  t.q[1] = *(const uint4*)(p + 16);
  return t.v;
}

static __device__ __forceinline__ v8f wmma_bf16(v16bf a, v16bf b, v8f c) {
  return __builtin_amdgcn_wmma_f32_16x16x32_bf16(false, a, false, b, (short)0, c, false, false);
}

// TDM 2D tile load: 64x64 bf16 tile, rows contiguous (stride 64 elems), into LDS.
// D# bitfields per CDNA5 ISA 8.3/8.4 (group0: count/lds/global/type, group1: dims).
static __device__ __forceinline__ void tdm_load_64x64_bf16(const u16* gsrc, u32 lds_off) {
  const u64 ga = (u64)(uintptr_t)gsrc;
  v4u g0;
  g0.x = 1u;                                            // count=1, user descriptor
  g0.y = lds_off;                                       // LDS byte address
  g0.z = (u32)(ga & 0xFFFFFFFFu);                       // global_addr[31:0]
  g0.w = (u32)((ga >> 32) & 0x01FFFFFFu) | (2u << 30);  // global_addr[56:32] | type=2
  const v8i g1 = {
    0x00010000,          // data_size=1 (2 bytes); wg_mask=0; no pad/iterate
    (int)(64u << 16),    // tensor_dim0 = 64  (bits [79:48], low 16)
    (int)(64u << 16),    // tensor_dim0 hi=0; tensor_dim1 = 64 (bits [111:80], low 16)
    (int)(64u << 16),    // tensor_dim1 hi=0; tile_dim0 = 64 (bits [127:112])
    64,                  // tile_dim1 = 64; tile_dim2 = 0
    64,                  // tensor_dim0_stride = 64 (low 32 of bits [207:160])
    0, 0                 // stride0 hi, tensor_dim1_stride = 0 (unused for 2D)
  };
  const v4i gz = {0, 0, 0, 0};
#if defined(__clang_major__) && (__clang_major__ >= 23)
  const v8i gz8 = {0, 0, 0, 0, 0, 0, 0, 0};
  __builtin_amdgcn_tensor_load_to_lds(g0, g1, gz, gz, gz8, 0);
#else
  __builtin_amdgcn_tensor_load_to_lds(g0, g1, gz, gz, 0);
#endif
}

// ---------------------------------------------------------------------------
// Pack kernels
// ---------------------------------------------------------------------------
__global__ void k_f32_to_bf16(const float* __restrict__ src, u16* __restrict__ dst, int n) {
  int i = (blockIdx.x * blockDim.x + threadIdx.x) * 8;
  if (i + 8 > n) return;
  float4 a = *(const float4*)(src + i);
  float4 b = *(const float4*)(src + i + 4);
  uint4 o;
  o.x = (u32)f2bf(a.x) | ((u32)f2bf(a.y) << 16);
  o.y = (u32)f2bf(a.z) | ((u32)f2bf(a.w) << 16);
  o.z = (u32)f2bf(b.x) | ((u32)f2bf(b.y) << 16);
  o.w = (u32)f2bf(b.z) | ((u32)f2bf(b.w) << 16);
  *(uint4*)(dst + i) = o;
}

// Fused, transposed QKV weight [640][512] + fused bias [640]
__global__ void k_pack_qkv_w(const float* __restrict__ Wk, const float* __restrict__ Wv,
                             const float* __restrict__ Wq, const float* __restrict__ bk,
                             const float* __restrict__ bv, const float* __restrict__ bq,
                             u16* __restrict__ Wt, float* __restrict__ bias) {
  int tid = blockIdx.x * blockDim.x + threadIdx.x;
  if (tid >= NQKV * WW) return;
  int n = tid >> 9;        // output column 0..639
  int k = tid & (WW - 1);  // input channel 0..511
  float w;
  if (n < DD)            w = Wk[k * DD + n];
  else if (n < 2 * DD)   w = Wv[k * DD + (n - DD)];
  else { int nn = n - 2 * DD; int h = nn >> 6, d = nn & 63; w = Wq[((size_t)h * WW + k) * DD + d]; }
  Wt[(size_t)n * WW + k] = f2bf(w);
  if (tid < NQKV) {
    float bb;
    if (tid < DD)          bb = bk[tid];
    else if (tid < 2 * DD) bb = bv[tid - DD];
    else                   bb = bq[tid - 2 * DD];   // bq is [H,D] flat
    bias[tid] = bb;
  }
}

// Transposed out-proj weight: Wpt[d][c] = Wp[c][d], d<64, c<512
__global__ void k_pack_wp(const float* __restrict__ Wp, u16* __restrict__ Wpt) {
  int tid = blockIdx.x * blockDim.x + threadIdx.x;
  if (tid >= DD * WW) return;
  int d = tid >> 9;
  int c = tid & (WW - 1);
  Wpt[(size_t)d * WW + c] = f2bf(Wp[(size_t)c * DD + d]);
}

// ---------------------------------------------------------------------------
// Fused QKV projection: [8192,512] x [512,640] -> K,V (bf16) and Q (bf16, [b,h,s,d])
// 8 waves/block, 16 rows per wave, A-tile resident in VGPRs.
// ---------------------------------------------------------------------------
__global__ void __launch_bounds__(256) k_qkv_gemm(
    const u16* __restrict__ xb, const u16* __restrict__ Wt, const float* __restrict__ bias,
    u16* __restrict__ Kb, u16* __restrict__ Vb, u16* __restrict__ Qb) {
  const int lane = threadIdx.x & 31;
  const int wave = threadIdx.x >> 5;
  const int hf   = lane >> 4;
  const int nl   = lane & 15;
  const int row0 = (blockIdx.x * 8 + wave) * 16;
  const u16* arow = xb + (size_t)row0 * WW;

  v16bf a[16];
#pragma unroll
  for (int ki = 0; ki < 16; ++ki) a[ki] = load_a16x32(arow, WW, lane, ki * 32);

  for (int nt = 0; nt < NQKV / 16; ++nt) {
    v8f acc = {};
    const u16* wcol = Wt + (size_t)nt * 16 * WW;
#pragma unroll
    for (int ki = 0; ki < 16; ++ki) {
      v16bf b = load_b32x16(wcol, WW, lane, ki * 32);
      acc = wmma_bf16(a[ki], b, acc);
    }
    const int col = nt * 16 + nl;
    const float bb = bias[col];
#pragma unroll
    for (int r = 0; r < 8; ++r) {
      const int row = row0 + r + hf * 8;
      const u16 val = f2bf(acc[r] + bb);
      if (col < DD) {
        Kb[(size_t)row * DD + col] = val;
      } else if (col < 2 * DD) {
        Vb[(size_t)row * DD + (col - DD)] = val;
      } else {
        const int cc = col - 2 * DD;
        const int h = cc >> 6, d = cc & 63;
        const int bi = row >> 12;           // row / 4096
        const int s  = row & (SS - 1);
        Qb[(((size_t)bi * HH + h) * SS + s) * DD + d] = val;
      }
    }
  }
}

// ---------------------------------------------------------------------------
// Flash attention: block = (b,h, 128 query rows), 8 waves, 16 rows per wave.
// KV tiles (64 keys) staged by the Tensor Data Mover into LDS; V transposed
// LDS->LDS for P@V B-fragments. Online softmax uses half-wave shuffles.
// ---------------------------------------------------------------------------
__global__ void __launch_bounds__(256) k_attn(
    const u16* __restrict__ Qb, const u16* __restrict__ Kb,
    const u16* __restrict__ Vb, u16* __restrict__ Zc) {
  __shared__ u16 Ks[64 * 64];       // [key][d]   row-major (TDM destination)
  __shared__ u16 Vs[64 * 64];       // [key][d]   row-major (TDM destination)
  __shared__ u16 Vt[64 * 64];       // [d][key]   transposed
  __shared__ u16 Pl[8 * 16 * 64];   // per-wave P staging [16 rows][64 keys]

  const int lane = threadIdx.x & 31;
  const int wave = threadIdx.x >> 5;
  const int hf   = lane >> 4;
  const int nl   = lane & 15;
  const int bi = blockIdx.z;
  const int h  = blockIdx.y;
  const int q0 = blockIdx.x * 128 + wave * 16;

  const u16* qbase = Qb + (((size_t)bi * HH + h) * SS + q0) * DD;
  const v16bf qa0 = load_a16x32(qbase, DD, lane, 0);
  const v16bf qa1 = load_a16x32(qbase, DD, lane, 32);

  v8f z[4] = {v8f{}, v8f{}, v8f{}, v8f{}};
  float m[8], l[8];
#pragma unroll
  for (int r = 0; r < 8; ++r) { m[r] = -3.0e38f; l[r] = 0.0f; }

  const u16* Kg = Kb + (size_t)bi * SS * DD;
  const u16* Vg = Vb + (size_t)bi * SS * DD;
  u16* pw = Pl + wave * (16 * 64);
  const float scale = 0.125f;   // 1/sqrt(64)

  const u32 ks_off = (u32)(uintptr_t)(&Ks[0]);
  const u32 vs_off = (u32)(uintptr_t)(&Vs[0]);

  for (int kt = 0; kt < SS; kt += 64) {
    __syncthreads();   // previous iteration done reading Ks/Vt before TDM overwrite
    if (wave == 0) {
      tdm_load_64x64_bf16(Kg + (size_t)kt * DD, ks_off);
      tdm_load_64x64_bf16(Vg + (size_t)kt * DD, vs_off);
      __builtin_amdgcn_s_wait_tensorcnt(0);
    }
    if (kt + 64 < SS) {
      __builtin_prefetch(Kg + (size_t)(kt + 64) * DD + threadIdx.x * 16, 0, 0);
      __builtin_prefetch(Vg + (size_t)(kt + 64) * DD + threadIdx.x * 16, 0, 0);
    }
    __syncthreads();   // tiles visible to all waves

    // transpose V tile: thread owns key t0, d = d0..d0+15 (LDS -> regs -> LDS)
    {
      const int e0 = threadIdx.x * 16;
      const int t0 = e0 >> 6;
      const int d0 = e0 & 63;
      const uint4 va  = *(const uint4*)&Vs[e0];
      const uint4 vb4 = *(const uint4*)&Vs[e0 + 8];
      const u32 w[8] = {va.x, va.y, va.z, va.w, vb4.x, vb4.y, vb4.z, vb4.w};
#pragma unroll
      for (int j = 0; j < 8; ++j) {
        Vt[(d0 + 2 * j)     * 64 + t0] = (u16)(w[j] & 0xFFFFu);
        Vt[(d0 + 2 * j + 1) * 64 + t0] = (u16)(w[j] >> 16);
      }
    }
    __syncthreads();   // Vt ready

    // scores: S = Q @ K^T  (16 q-rows x 64 keys)
    v8f sf[4];
#pragma unroll
    for (int ns = 0; ns < 4; ++ns) {
      v8f sc = {};
      sc = wmma_bf16(qa0, load_b32x16(Ks + ns * 16 * 64, 64, lane, 0), sc);
      sc = wmma_bf16(qa1, load_b32x16(Ks + ns * 16 * 64, 64, lane, 32), sc);
      sf[ns] = sc;
    }

    // online softmax over this 64-key chunk; write P (bf16) to wave-private LDS
#pragma unroll
    for (int r = 0; r < 8; ++r) {
      float v0 = sf[0][r] * scale, v1 = sf[1][r] * scale;
      float v2 = sf[2][r] * scale, v3 = sf[3][r] * scale;
      float mx = fmaxf(fmaxf(v0, v1), fmaxf(v2, v3));
#pragma unroll
      for (int msk = 1; msk < 16; msk <<= 1) mx = fmaxf(mx, __shfl_xor(mx, msk, 32));
      const float mn = fmaxf(m[r], mx);
      const float al = __expf(m[r] - mn);
      m[r] = mn;
      const float p0 = __expf(v0 - mn), p1 = __expf(v1 - mn);
      const float p2 = __expf(v2 - mn), p3 = __expf(v3 - mn);
      float sum = (p0 + p1) + (p2 + p3);
#pragma unroll
      for (int msk = 1; msk < 16; msk <<= 1) sum += __shfl_xor(sum, msk, 32);
      l[r] = l[r] * al + sum;
      z[0][r] *= al; z[1][r] *= al; z[2][r] *= al; z[3][r] *= al;
      const int prow = r + hf * 8;
      u16* pp = pw + prow * 64 + nl;
      pp[0]  = f2bf(p0);
      pp[16] = f2bf(p1);
      pp[32] = f2bf(p2);
      pp[48] = f2bf(p3);
    }
    // DS ops are in-order within a wave: P stores precede the A-frag loads below.

    // Z += P @ V  (16 x 64) x (64 x 64)
#pragma unroll
    for (int kk = 0; kk < 2; ++kk) {
      const v16bf pa = load_a16x32(pw, 64, lane, kk * 32);
#pragma unroll
      for (int nd = 0; nd < 4; ++nd) {
        z[nd] = wmma_bf16(pa, load_b32x16(Vt + nd * 16 * 64, 64, lane, kk * 32), z[nd]);
      }
    }
  }

  // normalize and emit concat-head layout Zc[b, s, h*64 + d] as bf16
  float rinv[8];
#pragma unroll
  for (int r = 0; r < 8; ++r) rinv[r] = 1.0f / l[r];
#pragma unroll
  for (int nd = 0; nd < 4; ++nd) {
#pragma unroll
    for (int r = 0; r < 8; ++r) {
      const int srow = q0 + r + hf * 8;
      const int col  = h * DD + nd * 16 + nl;
      Zc[((size_t)bi * SS + srow) * (HH * DD) + col] = f2bf(z[nd][r] * rinv[r]);
    }
  }
}

// ---------------------------------------------------------------------------
// Output projection: [8192,512] x [512,64] + bp -> f32 out
// ---------------------------------------------------------------------------
__global__ void __launch_bounds__(256) k_proj(
    const u16* __restrict__ Zc, const u16* __restrict__ Wpt,
    const float* __restrict__ bp, float* __restrict__ out) {
  const int lane = threadIdx.x & 31;
  const int wave = threadIdx.x >> 5;
  const int hf   = lane >> 4;
  const int nl   = lane & 15;
  const int row0 = (blockIdx.x * 8 + wave) * 16;
  const u16* arow = Zc + (size_t)row0 * (HH * DD);

  v8f acc[4] = {v8f{}, v8f{}, v8f{}, v8f{}};
#pragma unroll 4
  for (int ki = 0; ki < 16; ++ki) {
    const v16bf a = load_a16x32(arow, HH * DD, lane, ki * 32);
#pragma unroll
    for (int nd = 0; nd < 4; ++nd) {
      acc[nd] = wmma_bf16(a, load_b32x16(Wpt + (size_t)nd * 16 * WW, WW, lane, ki * 32), acc[nd]);
    }
  }
#pragma unroll
  for (int nd = 0; nd < 4; ++nd) {
    const int col = nd * 16 + nl;
    const float bb = bp[col];
#pragma unroll
    for (int r = 0; r < 8; ++r) {
      const int row = row0 + r + hf * 8;
      out[(size_t)row * DD + col] = acc[nd][r] + bb;
    }
  }
}

// ---------------------------------------------------------------------------
// Host launcher. Workspace layout (bytes):
//   xb    @ 0        : 8192*512 bf16           (8 MiB)
//   Wqkvt @ 8388608  : 640*512 bf16
//   Wpt   @ 9043968  : 64*512 bf16
//   bqkv  @ 9109504  : 640 f32
//   Kb    @ 9112064  : 8192*64 bf16
//   Vb    @ 10160640 : 8192*64 bf16
//   Qb    @ 11209216 : 2*8*4096*64 bf16        (8 MiB)
//   Zc    @ 19597824 : 8192*512 bf16           (8 MiB)   total ~26.7 MiB
// ---------------------------------------------------------------------------
extern "C" void kernel_launch(void* const* d_in, const int* in_sizes, int n_in,
                              void* d_out, int out_size, void* d_ws, size_t ws_size,
                              hipStream_t stream) {
  (void)in_sizes; (void)n_in; (void)out_size; (void)ws_size;
  const float* x  = (const float*)d_in[0];
  const float* Wq = (const float*)d_in[1];
  const float* bq = (const float*)d_in[2];
  const float* Wk = (const float*)d_in[3];
  const float* bk = (const float*)d_in[4];
  const float* Wv = (const float*)d_in[5];
  const float* bv = (const float*)d_in[6];
  const float* Wp = (const float*)d_in[7];
  const float* bp = (const float*)d_in[8];
  float* out = (float*)d_out;

  char* ws = (char*)d_ws;
  u16*   xb    = (u16*)  (ws + 0);
  u16*   Wqkvt = (u16*)  (ws + 8388608);
  u16*   Wpt   = (u16*)  (ws + 9043968);
  float* bqkv  = (float*)(ws + 9109504);
  u16*   Kb    = (u16*)  (ws + 9112064);
  u16*   Vb    = (u16*)  (ws + 10160640);
  u16*   Qb    = (u16*)  (ws + 11209216);
  u16*   Zc    = (u16*)  (ws + 19597824);

  k_f32_to_bf16<<<(BS * WW / 8 + 255) / 256, 256, 0, stream>>>(x, xb, BS * WW);
  k_pack_qkv_w<<<(NQKV * WW + 255) / 256, 256, 0, stream>>>(Wk, Wv, Wq, bk, bv, bq, Wqkvt, bqkv);
  k_pack_wp<<<(DD * WW + 255) / 256, 256, 0, stream>>>(Wp, Wpt);

  k_qkv_gemm<<<BS / 128, 256, 0, stream>>>(xb, Wqkvt, bqkv, Kb, Vb, Qb);

  dim3 ag(SS / 128, HH, BB);
  k_attn<<<ag, 256, 0, stream>>>(Qb, Kb, Vb, Zc);

  k_proj<<<BS / 128, 256, 0, stream>>>(Zc, Wpt, bp, out);
}
